// TprRnn_80367428043378
// MI455X (gfx1250) — matrix operand order repaired and checked by reading, not compiled, not run on hardware
//
#include <hip/hip_runtime.h>
#include <hip/hip_bf16.h>
#include <math.h>

// ---------------- model constants ----------------
#define B_ 64
#define S_ 50
#define W_ 12
#define SYMD 128
#define SLOTD 128
#define ENTD 128
#define ROLED 32
#define MLPD 256
#define NB_ 5
#define NR_ 4
#define VV 50000
#define EPSF 1e-6f

// input indices (setup_inputs dict order, params flattened in insertion order)
enum {
  IN_STORY=0, IN_QUERY=1, IN_WE=2, IN_POS=3,
  IN_WQ=4, IN_BQ=5, IN_WK=6, IN_BK=7, IN_WV=8, IN_BV=9,
  IN_W1=10, IN_B1=11, IN_W2=12, IN_B2=13, IN_GPRE=14, IN_BPRE=15,
  IN_WB=16, IN_BB=17, IN_WR=18, IN_BR=19, IN_WP=20, IN_BP=21,
  IN_UWE1=22, IN_UBE1=23, IN_UWE2=24, IN_UBE2=25,
  IN_UWR1=26, IN_UBR1=27, IN_UWR2=28, IN_UBR2=29, IN_UWR3=30, IN_UBR3=31,
  IN_IWE1=32, IN_IBE1=33, IN_IWR1=34, IN_IBR1=35, IN_IWR2=36, IN_IBR2=37,
  IN_IWR3=38, IN_IBR3=39, IN_ZW=40, IN_ZB=41,
  IN_G1=42, IN_BB1=43, IN_G2=44, IN_BB2=45, IN_G3=46, IN_BB3=47
};

typedef __attribute__((ext_vector_type(16))) __bf16 v16bf;
typedef __attribute__((ext_vector_type(8)))  float  v8f;

// epilogues for the generic WMMA GEMM
#define EPI_NONE 0
#define EPI_ELU1 1      // elu(x)+1
#define EPI_RELU 2
#define EPI_TANH 3
#define EPI_ADDSCALE 4  // Y += x*scale
#define EPI_QELU 5      // elu((x + extra)*scale)+1

#define MT 4            // M-tiles per wave (register blocking; B fragment reused 4x)

// ---------------------------------------------------------------
// Y[M,N] = epi(X[M,K] @ W[N,K]^T + bias), bf16 WMMA, f32 accumulate.
// One wave (32 threads) computes a 64x16 block: MT=4 accumulators share
// each B fragment, giving 4 back-to-back v_wmma per K-step.
// grid = (M/64, N/16). Fragment layouts per CDNA5 ISA 7.12.2.
// ---------------------------------------------------------------
__global__ __launch_bounds__(32)
void gemm_bf16_kernel(const float* __restrict__ X, const float* __restrict__ Wt,
                      const float* __restrict__ bias, const float* __restrict__ extra,
                      float* __restrict__ Y, int N, int K, int epi, float scale)
{
  const int lane = threadIdx.x;            // wave32 lane
  const int m0 = blockIdx.x * (16 * MT);
  const int n0 = blockIdx.y * 16;

  const int kb = (lane >> 4) * 8;          // A: K sub-offset per lane half
  const int ks = (lane >> 4) * 8;          // B: K sub-offset per lane half

  const float* __restrict__ ap[MT];
#pragma unroll
  for (int t = 0; t < MT; ++t)
    ap[t] = X + (size_t)(m0 + t * 16 + (lane & 15)) * K;
  const float* __restrict__ bp = Wt + (size_t)(n0 + (lane & 15)) * K;

  v8f acc[MT];
#pragma unroll
  for (int t = 0; t < MT; ++t) acc[t] = (v8f){0.f,0.f,0.f,0.f,0.f,0.f,0.f,0.f};

  for (int kc = 0; kc < K; kc += 32) {
    if (kc + 32 < K) {                       // global_prefetch_b8 for next chunk
      __builtin_prefetch(bp + kc + 32, 0, 0);
#pragma unroll
      for (int t = 0; t < MT; ++t) __builtin_prefetch(ap[t] + kc + 32, 0, 0);
    }
    v16bf b;
#pragma unroll
    for (int j = 0; j < 8; ++j) {
      b[2*j]     = (__bf16)bp[kc + ks + 2*j + (j & 4) * 2];
      b[2*j + 1] = (__bf16)bp[kc + ks + 2*j + (j & 4) * 2 + 1];
    }
#pragma unroll
    for (int t = 0; t < MT; ++t) {
      v16bf a;
#pragma unroll
      for (int j = 0; j < 4; ++j) {
        a[2*j]         = (__bf16)ap[t][kc + kb + 2*j];
        a[2*j + 1]     = (__bf16)ap[t][kc + kb + 2*j + 1];
        a[8 + 2*j]     = (__bf16)ap[t][kc + 16 + kb + 2*j];
        a[8 + 2*j + 1] = (__bf16)ap[t][kc + 16 + kb + 2*j + 1];
      }
      acc[t] = __builtin_amdgcn_wmma_f32_16x16x32_bf16(false, a, false, b,
                                                       (short)0, acc[t], false, false);
    }
  }

  // C/D layout: VGPR j -> M = j (lanes 0-15) or j+8 (lanes 16-31), N = lane&15
  const int nn = n0 + (lane & 15);
  const float bv = bias ? bias[nn] : 0.f;
#pragma unroll
  for (int t = 0; t < MT; ++t) {
    const int mb = m0 + t * 16 + ((lane >> 4) * 8);
#pragma unroll
    for (int j = 0; j < 8; ++j) {
      const size_t oi = (size_t)(mb + j) * N + nn;
      float v = acc[t][j] + bv;
      switch (epi) {
        case EPI_NONE: Y[oi] = v; break;
        case EPI_ELU1: Y[oi] = (v > 0.f) ? (v + 1.f) : __expf(v); break;
        case EPI_RELU: Y[oi] = v > 0.f ? v : 0.f; break;
        case EPI_TANH: Y[oi] = tanhf(v); break;
        case EPI_ADDSCALE: Y[oi] += v * scale; break;
        case EPI_QELU: {
          float q = (v + extra[oi]) * scale;
          Y[oi] = (q > 0.f) ? (q + 1.f) : __expf(q);
        } break;
      }
    }
  }
}

// ---------------- embedding + masked sums ----------------
__global__ void embed_kernel(const int* __restrict__ tok, const float* __restrict__ we,
                             const float* __restrict__ pos, float* __restrict__ sent,
                             float* __restrict__ ssum)
{
  const int seq = blockIdx.x;      // sentence index
  const int e = threadIdx.x;       // 0..127
  float acc = 0.f;
  for (int w = 0; w < W_; ++w) {
    const int t = tok[seq * W_ + w];
    const float v = we[(size_t)t * SYMD + e] * pos[w * SYMD + e];
    sent[((size_t)seq * W_ + w) * SYMD + e] = v;
    if (t != 0) acc += v;
  }
  ssum[(size_t)seq * SYMD + e] = acc;
}

// ---------------- slot-attention core (per sentence) ----------------
__global__ __launch_bounds__(128)
void attn_core_kernel(const float* __restrict__ kbuf, const float* __restrict__ vbuf,
                      const float* __restrict__ qbuf, const int* __restrict__ tok,
                      const float* __restrict__ gpre, const float* __restrict__ bpre,
                      float* __restrict__ hout, int nslots)
{
  const int seq = blockIdx.x;
  const int tid = threadIdx.x;
  __shared__ float sk[W_][SLOTD];
  __shared__ float sv[W_][SLOTD];
  __shared__ float sq[NB_][SLOTD];
  __shared__ float slog[W_][NB_];
  __shared__ float scol[NB_];
  __shared__ float sred[128];

  for (int i = tid; i < W_ * SLOTD; i += 128) {
    sk[i >> 7][i & 127] = kbuf[(size_t)seq * W_ * SLOTD + i];
    sv[i >> 7][i & 127] = vbuf[(size_t)seq * W_ * SLOTD + i];
  }
  for (int i = tid; i < nslots * SLOTD; i += 128)
    sq[i >> 7][i & 127] = qbuf[(size_t)seq * nslots * SLOTD + i];
  __syncthreads();

  // logits[w][n] = k_w . q_n + madd
  if (tid < W_ * nslots) {
    const int w = tid / nslots, n = tid % nslots;
    float acc = 0.f;
    for (int d = 0; d < SLOTD; ++d) acc += sk[w][d] * sq[n][d];
    if (tok[seq * W_ + w] == 0) acc += -1e6f;
    slog[w][n] = acc;
  }
  __syncthreads();
  // softmax over slots + EPS
  if (tid < W_) {
    float mx = -1e30f;
    for (int n = 0; n < nslots; ++n) mx = fmaxf(mx, slog[tid][n]);
    float s = 0.f; float ex[NB_];
    for (int n = 0; n < nslots; ++n) { ex[n] = __expf(slog[tid][n] - mx); s += ex[n]; }
    for (int n = 0; n < nslots; ++n) slog[tid][n] = ex[n] / s + EPSF;
  }
  __syncthreads();
  if (tid < nslots) {
    float cs = 0.f;
    for (int w = 0; w < W_; ++w) cs += slog[w][tid];
    scol[tid] = cs;
  }
  __syncthreads();
  if (tid < W_ * nslots) {
    const int w = tid / nslots, n = tid % nslots;
    slog[w][n] /= scol[n];
  }
  __syncthreads();

  // upd + LayerNorm(eps=1e-5) per slot row
  for (int n = 0; n < nslots; ++n) {
    float u = 0.f;
    for (int w = 0; w < W_; ++w) u += slog[w][n] * sv[w][tid];
    sred[tid] = u; __syncthreads();
    for (int st = 64; st > 0; st >>= 1) { if (tid < st) sred[tid] += sred[tid + st]; __syncthreads(); }
    const float mean = sred[0] * (1.f / 128.f); __syncthreads();
    const float d = u - mean;
    sred[tid] = d * d; __syncthreads();
    for (int st = 64; st > 0; st >>= 1) { if (tid < st) sred[tid] += sred[tid + st]; __syncthreads(); }
    const float var = sred[0] * (1.f / 128.f); __syncthreads();
    hout[((size_t)seq * nslots + n) * SLOTD + tid] =
        d * rsqrtf(var + 1e-5f) * gpre[tid] + bpre[tid];
  }
}

// ---------------- concat [init | slots] -> [rows,256] ----------------
__global__ void concat_kernel(const float* __restrict__ a, const float* __restrict__ b,
                              float* __restrict__ y, int rows)
{
  const size_t i = (size_t)blockIdx.x * 256 + threadIdx.x;
  if (i >= (size_t)rows * 256) return;
  const size_t row = i >> 8; const int col = (int)(i & 255);
  y[i] = (col < 128) ? a[row * 128 + col] : b[row * 128 + (col - 128)];
}

// ---------------- slot selector mixing (roll(am,r)) ----------------
__global__ void selmix_kernel(const float* __restrict__ slots, float* __restrict__ sel,
                              int nslots, int rows, float big)
{
  const int row = blockIdx.x;
  const int e = threadIdx.x;   // 0..127
  float sv[NB_];
  for (int n = 0; n < nslots; ++n)
    sv[n] = slots[((size_t)row * nslots + n) * SLOTD + e];
  for (int r = 0; r < nslots; ++r) {
    float acc = 0.f;
    for (int n = 0; n < nslots; ++n) acc += ((n == r) ? big : EPSF) * sv[n];
    sel[((size_t)r * rows + row) * SLOTD + e] = acc;
  }
}

__global__ void zero_kernel(float* __restrict__ p, size_t n)
{
  const size_t i = (size_t)blockIdx.x * blockDim.x + threadIdx.x;
  if (i < n) p[i] = 0.f;
}

// ---------------- TPR scan: hats (pass 1) ----------------
// grid (B, 8 e-chunks), 128 threads (f). part[b][c][3][128]
__global__ __launch_bounds__(128)
void tpr_hats_kernel(const float* __restrict__ T, const float* __restrict__ e1b,
                     const float* __restrict__ e2b, const float* __restrict__ r1b,
                     const float* __restrict__ r2b, const float* __restrict__ r3b,
                     float* __restrict__ part, int s)
{
  const int b = blockIdx.x, c = blockIdx.y, f = threadIdx.x;
  const int row = b * S_ + s;
  __shared__ float sr1[ROLED], sr2[ROLED], sr3[ROLED], se1[16], se2[16];
  if (f < ROLED) { sr1[f] = r1b[(size_t)row*ROLED+f]; sr2[f] = r2b[(size_t)row*ROLED+f]; sr3[f] = r3b[(size_t)row*ROLED+f]; }
  if (f < 16)    { se1[f] = e1b[(size_t)row*ENTD + c*16 + f]; se2[f] = e2b[(size_t)row*ENTD + c*16 + f]; }
  __syncthreads();
  float aw = 0.f, am = 0.f, ab = 0.f;
  const float* Tb = T + (size_t)b * ENTD * ROLED * ENTD;
  for (int ei = 0; ei < 16; ++ei) {
    const float* Te = Tb + (size_t)(c*16 + ei) * ROLED * ENTD + f;
    float s1 = 0.f, s2 = 0.f, s3 = 0.f;
#pragma unroll
    for (int r = 0; r < ROLED; ++r) {
      const float t = Te[r * ENTD];
      s1 += sr1[r] * t; s2 += sr2[r] * t; s3 += sr3[r] * t;
    }
    aw += se1[ei] * s1; am += se1[ei] * s2; ab += se2[ei] * s3;
  }
  float* pp = part + ((size_t)(b*8 + c) * 3) * ENTD;
  pp[f] = aw; pp[ENTD + f] = am; pp[2*ENTD + f] = ab;
}

// ---------------- TPR scan: rank-2 update (pass 2) ----------------
__global__ __launch_bounds__(128)
void tpr_update_kernel(float* __restrict__ T, const float* __restrict__ e1b,
                       const float* __restrict__ e2b, const float* __restrict__ r1b,
                       const float* __restrict__ r2b, const float* __restrict__ r3b,
                       const float* __restrict__ part, int s)
{
  const int b = blockIdx.x, c = blockIdx.y, f = threadIdx.x;
  const int row = b * S_ + s;
  float wh = 0.f, mh = 0.f, bh = 0.f;
  for (int cc = 0; cc < 8; ++cc) {
    const float* pp = part + ((size_t)(b*8 + cc) * 3) * ENTD;
    wh += pp[f]; mh += pp[ENTD + f]; bh += pp[2*ENTD + f];
  }
  __shared__ float su[ROLED][ENTD];   // write_op + move_op
  __shared__ float sbk[ROLED][ENTD];  // back_op
  __shared__ float sr1[ROLED], sr2[ROLED], sr3[ROLED];
  if (f < ROLED) { sr1[f] = r1b[(size_t)row*ROLED+f]; sr2[f] = r2b[(size_t)row*ROLED+f]; sr3[f] = r3b[(size_t)row*ROLED+f]; }
  __syncthreads();
  const float e1f = e1b[(size_t)row*ENTD + f];
  const float e2f = e2b[(size_t)row*ENTD + f];
  for (int r = 0; r < ROLED; ++r) {
    su[r][f]  = sr1[r] * (e2f - wh) + sr2[r] * (wh - mh);
    sbk[r][f] = sr3[r] * (e1f - bh);
  }
  __syncthreads();
  float* Tb = T + (size_t)b * ENTD * ROLED * ENTD;
  for (int ei = 0; ei < 16; ++ei) {
    const int e = c*16 + ei;
    const float ca = e1b[(size_t)row*ENTD + e];
    const float cb = e2b[(size_t)row*ENTD + e];
    float* Te = Tb + (size_t)e * ROLED * ENTD + f;
#pragma unroll
    for (int r = 0; r < ROLED; ++r)
      Te[r * ENTD] += ca * su[r][f] + cb * sbk[r][f];
  }
}

// ---------------- inference contractions ----------------
__global__ __launch_bounds__(128)
void infer_partial_kernel(const float* __restrict__ T, const float* __restrict__ probe,
                          const float* __restrict__ rr, float* __restrict__ part)
{
  const int b = blockIdx.x, c = blockIdx.y, f = threadIdx.x;
  __shared__ float sr[ROLED], se[16];
  if (f < ROLED) sr[f] = rr[(size_t)b*ROLED + f];
  if (f < 16)    se[f] = probe[(size_t)b*ENTD + c*16 + f];
  __syncthreads();
  float acc = 0.f;
  const float* Tb = T + (size_t)b * ENTD * ROLED * ENTD;
  for (int ei = 0; ei < 16; ++ei) {
    const float* Te = Tb + (size_t)(c*16 + ei) * ROLED * ENTD + f;
    float s1 = 0.f;
#pragma unroll
    for (int r = 0; r < ROLED; ++r) s1 += sr[r] * Te[r * ENTD];
    acc += se[ei] * s1;
  }
  part[((size_t)b*8 + c) * ENTD + f] = acc;
}

__global__ __launch_bounds__(128)
void infer_finish_kernel(const float* __restrict__ part, const float* __restrict__ g,
                         const float* __restrict__ bv, float* __restrict__ iout,
                         float* __restrict__ isum, int first)
{
  const int b = blockIdx.x, f = threadIdx.x;
  float h = 0.f;
  for (int c = 0; c < 8; ++c) h += part[((size_t)b*8 + c) * ENTD + f];
  __shared__ float sred[128];
  sred[f] = h; __syncthreads();
  for (int st = 64; st > 0; st >>= 1) { if (f < st) sred[f] += sred[f + st]; __syncthreads(); }
  const float mean = sred[0] * (1.f / 128.f); __syncthreads();
  const float d = h - mean;
  sred[f] = d * d; __syncthreads();
  for (int st = 64; st > 0; st >>= 1) { if (f < st) sred[f] += sred[f + st]; __syncthreads(); }
  const float var = sred[0] * (1.f / 128.f); __syncthreads();
  const float o = d * rsqrtf(var + 1e-6f) * g[f] + bv[f];
  iout[(size_t)b*ENTD + f] = o;
  if (first) isum[(size_t)b*ENTD + f] = o;
  else       isum[(size_t)b*ENTD + f] += o;
}

// ---------------------------------------------------------------
extern "C" void kernel_launch(void* const* d_in, const int* in_sizes, int n_in,
                              void* d_out, int out_size, void* d_ws, size_t ws_size,
                              hipStream_t stream)
{
  (void)in_sizes; (void)n_in; (void)out_size; (void)ws_size;
  const int*   story = (const int*)d_in[IN_STORY];
  const int*   query = (const int*)d_in[IN_QUERY];
  const float* we    = (const float*)d_in[IN_WE];
  const float* pos   = (const float*)d_in[IN_POS];
#define P(i) ((const float*)d_in[i])

  float* ws = (float*)d_ws;
  size_t off = 0;
  auto alloc = [&](size_t n) { float* p = ws + off; off += n; return p; };

  const int BS = B_ * S_;                 // 3200 sentences
  float* sent   = alloc((size_t)BS * W_ * SYMD);
  float* qemb   = alloc((size_t)B_ * W_ * SYMD);
  float* ssum   = alloc((size_t)BS * SYMD);
  float* qsum   = alloc((size_t)B_ * SYMD);
  float* kS     = alloc((size_t)BS * W_ * SLOTD);
  float* vS     = alloc((size_t)BS * W_ * SLOTD);
  float* kQ     = alloc((size_t)B_ * W_ * SLOTD);
  float* vQ     = alloc((size_t)B_ * W_ * SLOTD);
  float* initS  = alloc((size_t)BS * NB_ * SLOTD);   // [3200,640] == [16000,128]
  float* slotS  = alloc((size_t)BS * NB_ * SLOTD);
  float* qbufS  = alloc((size_t)BS * NB_ * SLOTD);
  float* hS     = alloc((size_t)BS * NB_ * SLOTD);
  float* t1S    = alloc((size_t)BS * NB_ * MLPD);
  float* catS   = alloc((size_t)BS * NB_ * 2 * SLOTD);
  float* bslots = alloc((size_t)BS * NB_ * SLOTD);
  float* initQ  = alloc((size_t)B_ * NR_ * SLOTD);
  float* slotQ  = alloc((size_t)B_ * NR_ * SLOTD);
  float* qbufQ  = alloc((size_t)B_ * NR_ * SLOTD);
  float* hQ     = alloc((size_t)B_ * NR_ * SLOTD);
  float* t1Q    = alloc((size_t)B_ * NR_ * MLPD);
  float* catQ   = alloc((size_t)B_ * NR_ * 2 * SLOTD);
  float* rslots = alloc((size_t)B_ * NR_ * SLOTD);
  float* kselS  = alloc((size_t)NB_ * BS * SLOTD);
  float* e1b    = alloc((size_t)BS * ENTD);
  float* e2b    = alloc((size_t)BS * ENTD);
  float* r1b    = alloc((size_t)BS * ROLED);
  float* r2b    = alloc((size_t)BS * ROLED);
  float* r3b    = alloc((size_t)BS * ROLED);
  float* TPR    = alloc((size_t)B_ * ENTD * ROLED * ENTD);  // 134 MB (L2-resident)
  float* hpart  = alloc((size_t)B_ * 8 * 3 * ENTD);
  float* qsel   = alloc((size_t)NR_ * B_ * SLOTD);
  float* e1q    = alloc((size_t)B_ * ENTD);
  float* r1q    = alloc((size_t)B_ * ROLED);
  float* r2q    = alloc((size_t)B_ * ROLED);
  float* r3q    = alloc((size_t)B_ * ROLED);
  float* ipart  = alloc((size_t)B_ * 8 * ENTD);
  float* itA    = alloc((size_t)B_ * ENTD);
  float* itB    = alloc((size_t)B_ * ENTD);
  float* isum   = alloc((size_t)B_ * ENTD);

  const float qscale = 0.08838834764831845f;   // SLOT^-0.5
  const float mscale = 1.f / 128.f;            // 1/SLOT
  const dim3 wv(32, 1, 1);

  // 1) embeddings + masked sums
  embed_kernel<<<BS, 128, 0, stream>>>(story, we, pos, sent, ssum);
  embed_kernel<<<B_, 128, 0, stream>>>(query, we, pos, qemb, qsum);

  // 2) k = elu(sent@Wk^T+bk)+1 ; v = sent@Wv^T+bv  (38400 / 768 rows, 64 rows per wave)
  gemm_bf16_kernel<<<dim3(600, 8), wv, 0, stream>>>(sent, P(IN_WK), P(IN_BK), nullptr, kS, 128, 128, EPI_ELU1, 0.f);
  gemm_bf16_kernel<<<dim3(600, 8), wv, 0, stream>>>(sent, P(IN_WV), P(IN_BV), nullptr, vS, 128, 128, EPI_NONE, 0.f);
  gemm_bf16_kernel<<<dim3(12, 8),  wv, 0, stream>>>(qemb, P(IN_WK), P(IN_BK), nullptr, kQ, 128, 128, EPI_ELU1, 0.f);
  gemm_bf16_kernel<<<dim3(12, 8),  wv, 0, stream>>>(qemb, P(IN_WV), P(IN_BV), nullptr, vQ, 128, 128, EPI_NONE, 0.f);

  // 3) init slots: ssum@Wb^T (N=640), qsum@Wr^T (N=512)
  gemm_bf16_kernel<<<dim3(50, 40), wv, 0, stream>>>(ssum, P(IN_WB), P(IN_BB), nullptr, initS, 640, 128, EPI_NONE, 0.f);
  gemm_bf16_kernel<<<dim3(1, 32),  wv, 0, stream>>>(qsum, P(IN_WR), P(IN_BR), nullptr, initQ, 512, 128, EPI_NONE, 0.f);
  hipMemcpyAsync(slotS, initS, (size_t)BS * NB_ * SLOTD * sizeof(float), hipMemcpyDeviceToDevice, stream);
  hipMemcpyAsync(slotQ, initQ, (size_t)B_ * NR_ * SLOTD * sizeof(float), hipMemcpyDeviceToDevice, stream);

  // 4) slot-attention iterations (story: 16000 slot rows; query: 256)
  for (int it = 0; it < 3; ++it) {
    gemm_bf16_kernel<<<dim3(250, 8), wv, 0, stream>>>(slotS, P(IN_WQ), P(IN_BQ), initS, qbufS, 128, 128, EPI_QELU, qscale);
    attn_core_kernel<<<BS, 128, 0, stream>>>(kS, vS, qbufS, story, P(IN_GPRE), P(IN_BPRE), hS, NB_);
    gemm_bf16_kernel<<<dim3(250, 16), wv, 0, stream>>>(hS, P(IN_W1), P(IN_B1), nullptr, t1S, 256, 128, EPI_RELU, 0.f);
    gemm_bf16_kernel<<<dim3(250, 8),  wv, 0, stream>>>(t1S, P(IN_W2), P(IN_B2), nullptr, slotS, 128, 256, EPI_ADDSCALE, mscale);
  }
  for (int it = 0; it < 3; ++it) {
    gemm_bf16_kernel<<<dim3(4, 8),  wv, 0, stream>>>(slotQ, P(IN_WQ), P(IN_BQ), initQ, qbufQ, 128, 128, EPI_QELU, qscale);
    attn_core_kernel<<<B_, 128, 0, stream>>>(kQ, vQ, qbufQ, query, P(IN_GPRE), P(IN_BPRE), hQ, NR_);
    gemm_bf16_kernel<<<dim3(4, 16), wv, 0, stream>>>(hQ, P(IN_W1), P(IN_B1), nullptr, t1Q, 256, 128, EPI_RELU, 0.f);
    gemm_bf16_kernel<<<dim3(4, 8),  wv, 0, stream>>>(t1Q, P(IN_W2), P(IN_B2), nullptr, slotQ, 128, 256, EPI_ADDSCALE, mscale);
  }

  // 5) concat + Wp projection
  concat_kernel<<<(BS * NB_ * 256 + 255) / 256, 256, 0, stream>>>(initS, slotS, catS, BS * NB_);
  gemm_bf16_kernel<<<dim3(250, 8), wv, 0, stream>>>(catS, P(IN_WP), P(IN_BP), nullptr, bslots, 128, 256, EPI_NONE, 0.f);
  concat_kernel<<<(B_ * NR_ * 256 + 255) / 256, 256, 0, stream>>>(initQ, slotQ, catQ, B_ * NR_);
  gemm_bf16_kernel<<<dim3(4, 8), wv, 0, stream>>>(catQ, P(IN_WP), P(IN_BP), nullptr, rslots, 128, 256, EPI_NONE, 0.f);

  // 6) selector mixing + update projections (tanh)
  selmix_kernel<<<BS, 128, 0, stream>>>(bslots, kselS, NB_, BS, 1.f - 4.f * EPSF);
  selmix_kernel<<<B_, 128, 0, stream>>>(rslots, qsel, NR_, B_, 1.f - 3.f * EPSF);
  gemm_bf16_kernel<<<dim3(50, 8), wv, 0, stream>>>(kselS + 0*(size_t)BS*SLOTD, P(IN_UWE1), P(IN_UBE1), nullptr, e1b, 128, 128, EPI_TANH, 0.f);
  gemm_bf16_kernel<<<dim3(50, 8), wv, 0, stream>>>(kselS + 1*(size_t)BS*SLOTD, P(IN_UWE2), P(IN_UBE2), nullptr, e2b, 128, 128, EPI_TANH, 0.f);
  gemm_bf16_kernel<<<dim3(50, 2), wv, 0, stream>>>(kselS + 2*(size_t)BS*SLOTD, P(IN_UWR1), P(IN_UBR1), nullptr, r1b, 32, 128, EPI_TANH, 0.f);
  gemm_bf16_kernel<<<dim3(50, 2), wv, 0, stream>>>(kselS + 3*(size_t)BS*SLOTD, P(IN_UWR2), P(IN_UBR2), nullptr, r2b, 32, 128, EPI_TANH, 0.f);
  gemm_bf16_kernel<<<dim3(50, 2), wv, 0, stream>>>(kselS + 4*(size_t)BS*SLOTD, P(IN_UWR3), P(IN_UBR3), nullptr, r3b, 32, 128, EPI_TANH, 0.f);

  // 7) TPR scan: zero, then 50 sequential (hats, update) passes (L2-resident tensor)
  {
    const size_t n = (size_t)B_ * ENTD * ROLED * ENTD;
    zero_kernel<<<(unsigned)((n + 255) / 256), 256, 0, stream>>>(TPR, n);
  }
  for (int s = 0; s < S_; ++s) {
    tpr_hats_kernel<<<dim3(B_, 8), 128, 0, stream>>>(TPR, e1b, e2b, r1b, r2b, r3b, hpart, s);
    tpr_update_kernel<<<dim3(B_, 8), 128, 0, stream>>>(TPR, e1b, e2b, r1b, r2b, r3b, hpart, s);
  }

  // 8) inference projections (no tanh) and chained TPR contractions with LN(1e-6)
  gemm_bf16_kernel<<<dim3(1, 8), wv, 0, stream>>>(qsel + 0*(size_t)B_*SLOTD, P(IN_IWE1), P(IN_IBE1), nullptr, e1q, 128, 128, EPI_NONE, 0.f);
  gemm_bf16_kernel<<<dim3(1, 2), wv, 0, stream>>>(qsel + 1*(size_t)B_*SLOTD, P(IN_IWR1), P(IN_IBR1), nullptr, r1q, 32, 128, EPI_NONE, 0.f);
  gemm_bf16_kernel<<<dim3(1, 2), wv, 0, stream>>>(qsel + 2*(size_t)B_*SLOTD, P(IN_IWR2), P(IN_IBR2), nullptr, r2q, 32, 128, EPI_NONE, 0.f);
  gemm_bf16_kernel<<<dim3(1, 2), wv, 0, stream>>>(qsel + 3*(size_t)B_*SLOTD, P(IN_IWR3), P(IN_IBR3), nullptr, r3q, 32, 128, EPI_NONE, 0.f);

  infer_partial_kernel<<<dim3(B_, 8), 128, 0, stream>>>(TPR, e1q, r1q, ipart);
  infer_finish_kernel<<<B_, 128, 0, stream>>>(ipart, P(IN_G1), P(IN_BB1), itA, isum, 1);
  infer_partial_kernel<<<dim3(B_, 8), 128, 0, stream>>>(TPR, itA, r2q, ipart);
  infer_finish_kernel<<<B_, 128, 0, stream>>>(ipart, P(IN_G2), P(IN_BB2), itB, isum, 0);
  infer_partial_kernel<<<dim3(B_, 8), 128, 0, stream>>>(TPR, itB, r3q, ipart);
  infer_finish_kernel<<<B_, 128, 0, stream>>>(ipart, P(IN_G3), P(IN_BB3), itA, isum, 0);

  // 9) logits: [64,128] @ ZW[50000,128]^T + Zb  (single M-block: ZW read exactly once)
  gemm_bf16_kernel<<<dim3(1, 3125), wv, 0, stream>>>(isum, P(IN_ZW), P(IN_ZB), nullptr, (float*)d_out, VV, 128, EPI_NONE, 0.f);
#undef P
}
